// PagedMinimalAttention_38680475468397
// MI455X (gfx1250) — matrix operand order repaired and checked by reading, not compiled, not run on hardware
//
#include <hip/hip_runtime.h>

typedef _Float16 half_t;
typedef __attribute__((ext_vector_type(16))) _Float16 v16h;
typedef __attribute__((ext_vector_type(8)))  _Float16 v8h;
typedef __attribute__((ext_vector_type(8)))  float    v8f;
typedef __attribute__((ext_vector_type(4)))  unsigned int u32x4;
typedef __attribute__((ext_vector_type(8)))  int      i32x8;
typedef __attribute__((ext_vector_type(4)))  int      i32x4;

#define EMBED 1024
#define NHEAD 16
#define HDIM  64
#define BATCH 4
#define SEQ   2048
#define MTOT  (BATCH*SEQ)   // 8192 rows
#define BH    (BATCH*NHEAD) // 64 batch-heads

#if __has_builtin(__builtin_amdgcn_tensor_load_to_lds)
#define HAVE_TDM 1
#else
#define HAVE_TDM 0
#endif

// ---------------------------------------------------------------------------
// WMMA fragment loaders (per CDNA5 ISA 7.12.2 layouts, wave32)
// A (16x32 f16): lane L holds row M=L&15; lanes<16 get K={0..7,16..23},
//                lanes>=16 get K={8..15,24..31}  -> two 16B loads.
// B (32x16 f16): lane L holds col N=L&15; lanes<16 get K={0..7,16..23},
//                lanes>=16 get K={8..15,24..31}  -> two 16B loads.
// C/D (16x16 f32): VGPR r, lane L -> element (M = r + 8*(L>>4), N = L&15).
// ---------------------------------------------------------------------------
static __device__ inline v16h load_a_frag(const half_t* __restrict__ base, int ld) {
  int lane = threadIdx.x & 31;
  int row  = lane & 15;
  int klo  = (lane >> 4) << 3;          // 0 or 8
  const half_t* p = base + (size_t)row * ld + klo;
  union { v16h v; v8h h[2]; } f;
  f.h[0] = *(const v8h*)(p);            // K = klo .. klo+7
  f.h[1] = *(const v8h*)(p + 16);       // K = 16+klo .. 16+klo+7
  return f.v;
}

// src stored row-major as [n][k]; element = base[n*ld + k]
static __device__ inline v16h load_b_frag(const half_t* __restrict__ base, int ld) {
  int lane = threadIdx.x & 31;
  int n    = lane & 15;
  int klo  = (lane >> 4) << 3;          // 0 or 8
  const half_t* p = base + (size_t)n * ld + klo;
  union { v16h v; v8h h[2]; } f;
  f.h[0] = *(const v8h*)(p);            // K = klo .. klo+7
  f.h[1] = *(const v8h*)(p + 16);       // K = 16+klo .. 16+klo+7
  return f.v;
}

static __device__ inline v8f wmma_f16(v16h a, v16h b, v8f c) {
  return __builtin_amdgcn_wmma_f32_16x16x32_f16(false, a, false, b, (short)0, c,
                                                false, false);
}

#if HAVE_TDM
// Tensor DMA: 2-D f16 tile 64(x) x 32(y) from a [SEQ x HDIM] tensor into LDS
// with 8-DWORD pad per 32 DWORDs (row pitch 80 halves). ISA 8.3/8.4.
static __device__ __forceinline__ void tdm_load_k32(const half_t* gsrc,
                                                    half_t* ldst) {
  unsigned long long ga = (unsigned long long)(uintptr_t)gsrc;
  unsigned lds_off = (unsigned)(uintptr_t)ldst;
  u32x4 g0;
  g0[0] = 1u;                                    // count=1, user descriptor
  g0[1] = lds_off;                               // lds_addr
  g0[2] = (unsigned)ga;                          // global_addr[31:0]
  g0[3] = (unsigned)(ga >> 32) | (2u << 30);     // addr[56:32] | type=2
  i32x8 g1;
  g1[0] = (1 << 16)         // data_size = 2 bytes
        | (1 << 20)         // pad_enable
        | (4 << 22)         // pad_interval: 2^(4+1) = 32 DWORDs
        | (7 << 25);        // pad_amount: 7+1 = 8 DWORDs (16 halves)
  g1[1] = (HDIM << 16);     // tensor_dim0 = 64
  g1[2] = (SEQ << 16);      // tensor_dim0 hi=0 | tensor_dim1 lo = 2048
  g1[3] = (HDIM << 16);     // tensor_dim1 hi=0 | tile_dim0 = 64
  g1[4] = 32;               // tile_dim1 = 32 | tile_dim2 = 0
  g1[5] = HDIM;             // tensor_dim0_stride lo32 = 64
  g1[6] = 0;
  g1[7] = 0;
  i32x4 gz = {0, 0, 0, 0};
#if __clang_major__ >= 23
  i32x8 gz8 = {0, 0, 0, 0, 0, 0, 0, 0};
  __builtin_amdgcn_tensor_load_to_lds(g0, g1, gz, gz, gz8, 0);
#else
  __builtin_amdgcn_tensor_load_to_lds(g0, g1, gz, gz, 0);
#endif
}
#endif

// ---------------------------------------------------------------------------
// 1) f32 -> f16 conversion
// ---------------------------------------------------------------------------
__global__ void k_f32_to_f16(const float* __restrict__ src,
                             half_t* __restrict__ dst, int n) {
  int i = blockIdx.x * blockDim.x + threadIdx.x;
  int stride = gridDim.x * blockDim.x;
  for (; i < n; i += stride) dst[i] = (half_t)src[i];
}

// ---------------------------------------------------------------------------
// GEMM core: 8 waves (4M x 2N), wave tile 64x64 (16 accumulators),
// block tile 256x128. 16 WMMAs per 8 fragment loads per K-step.
// ---------------------------------------------------------------------------
#define GEMM_BODY(X, W)                                                       \
  int lane = threadIdx.x & 31;                                                \
  int wave = threadIdx.x >> 5;                                                \
  int wm = wave >> 1, wn = wave & 1;                                          \
  int m0 = blockIdx.y * 256 + wm * 64;                                        \
  int n0 = blockIdx.x * 128 + wn * 64;                                        \
  v8f acc[4][4];                                                              \
  _Pragma("unroll") for (int i = 0; i < 4; ++i)                               \
  _Pragma("unroll") for (int j = 0; j < 4; ++j)                               \
  _Pragma("unroll") for (int e = 0; e < 8; ++e) acc[i][j][e] = 0.f;           \
  for (int k = 0; k < EMBED; k += 32) {                                       \
    if (k + 64 < EMBED) {                                                     \
      __builtin_prefetch(X + (size_t)m0 * EMBED + (k + 64), 0, 0);            \
      __builtin_prefetch(W + (size_t)n0 * EMBED + (k + 64), 0, 0);            \
    }                                                                         \
    v16h a[4], bf[4];                                                         \
    _Pragma("unroll") for (int i = 0; i < 4; ++i)                             \
      a[i] = load_a_frag(X + (size_t)(m0 + i * 16) * EMBED + k, EMBED);       \
    _Pragma("unroll") for (int j = 0; j < 4; ++j)                             \
      bf[j] = load_b_frag(W + (size_t)(n0 + j * 16) * EMBED + k, EMBED);      \
    _Pragma("unroll") for (int i = 0; i < 4; ++i)                             \
    _Pragma("unroll") for (int j = 0; j < 4; ++j)                             \
      acc[i][j] = wmma_f16(a[i], bf[j], acc[i][j]);                           \
  }                                                                           \
  int hi = lane >> 4, l15 = lane & 15;

// ---------------------------------------------------------------------------
// 2) QKV projection: dst[bh][s][d] = ((X @ W^T)[m][n] + bias[n]) * scale
// ---------------------------------------------------------------------------
__global__ __launch_bounds__(256)
void k_proj_qkv(const half_t* __restrict__ X, const half_t* __restrict__ W,
                const float* __restrict__ bias, half_t* __restrict__ dst,
                float scale) {
  GEMM_BODY(X, W)
#pragma unroll
  for (int i = 0; i < 4; ++i) {
#pragma unroll
    for (int j = 0; j < 4; ++j) {
      int n = n0 + j * 16 + l15;
      int h = n >> 6, d = n & 63;
      float bv = bias[n];
#pragma unroll
      for (int r = 0; r < 8; ++r) {
        int m = m0 + i * 16 + r + 8 * hi;
        int bb = m >> 11, s = m & 2047;
        size_t idx = (((size_t)(bb * NHEAD + h) * SEQ) + s) * HDIM + d;
        dst[idx] = (half_t)((acc[i][j][r] + bv) * scale);
      }
    }
  }
}

// ---------------------------------------------------------------------------
// 3) Output projection: out[m][n] = (A @ Wo^T)[m][n] + bo[n]   (f32 out)
// ---------------------------------------------------------------------------
__global__ __launch_bounds__(256)
void k_proj_out(const half_t* __restrict__ X, const half_t* __restrict__ W,
                const float* __restrict__ bias, float* __restrict__ dst) {
  GEMM_BODY(X, W)
#pragma unroll
  for (int i = 0; i < 4; ++i) {
#pragma unroll
    for (int j = 0; j < 4; ++j) {
      int n = n0 + j * 16 + l15;
      float bv = bias[n];
#pragma unroll
      for (int r = 0; r < 8; ++r) {
        int m = m0 + i * 16 + r + 8 * hi;
        dst[(size_t)m * EMBED + n] = acc[i][j][r] + bv;
      }
    }
  }
}

// ---------------------------------------------------------------------------
// 4) Causal flash attention per (b,h). Block = 128 q-rows (8 waves x 16 rows).
//    K chunks (32x64 f16) double-buffered via the Tensor Data Mover:
//    chunk i+1 is issued before computing chunk i; s_wait_tensorcnt 1 keeps
//    one DMA in flight. V staged transposed by the VALU path (overlaps DMA).
//    Online softmax; Q carries 1/sqrt(D).
// ---------------------------------------------------------------------------
__global__ __launch_bounds__(256)
void k_attn(const half_t* __restrict__ Q, const half_t* __restrict__ K,
            const half_t* __restrict__ V, half_t* __restrict__ O) {
  __shared__ __align__(16) half_t Ks[2][32][80];   // double-buffered, pitch 80
  __shared__ __align__(16) half_t VsT[64][40];     // [d][t], padded
  __shared__ __align__(16) half_t Pbuf[8][16][32]; // per-wave P tile

  int lane = threadIdx.x & 31;
  int wave = threadIdx.x >> 5;
  int hi = lane >> 4, l15 = lane & 15;

  int bh = blockIdx.y;
  int b = bh >> 4, h = bh & 15;
  int qblk = blockIdx.x * 128;
  int q0 = qblk + wave * 16;

  const half_t* qp = Q + (size_t)bh * SEQ * HDIM;
  const half_t* kp = K + (size_t)bh * SEQ * HDIM;
  const half_t* vp = V + (size_t)bh * SEQ * HDIM;

  v16h qf0 = load_a_frag(qp + (size_t)q0 * HDIM + 0, HDIM);
  v16h qf1 = load_a_frag(qp + (size_t)q0 * HDIM + 32, HDIM);

  v8f oacc[4];
#pragma unroll
  for (int j = 0; j < 4; ++j)
#pragma unroll
    for (int e = 0; e < 8; ++e) oacc[j][e] = 0.f;
  float mrun[8], lrun[8];
#pragma unroll
  for (int r = 0; r < 8; ++r) { mrun[r] = -__builtin_inff(); lrun[r] = 0.f; }

  int tid = threadIdx.x;
  int tRow = tid >> 3;         // 0..31 : kv row within chunk
  int dCol = (tid & 7) * 8;    // 0..56 : 8-halves column group

  int kvend = qblk + 128;      // causal upper bound for this block
  int cur = 0;

#if HAVE_TDM
  if (wave == 0) tdm_load_k32(kp, &Ks[0][0][0]);  // prologue: chunk 0
#endif

  for (int kv = 0; kv < kvend; kv += 32) {
    // retire all reads of the buffer the next DMA will overwrite
    // (and last iteration's VsT reads before restaging V)
    __syncthreads();

#if HAVE_TDM
    if (wave == 0 && kv + 32 < kvend)
      tdm_load_k32(kp + (size_t)(kv + 32) * HDIM, &Ks[cur ^ 1][0][0]);
#else
    { // fallback: manual K staging
      v8h kd = *(const v8h*)(kp + (size_t)(kv + tRow) * HDIM + dCol);
      *(v8h*)(&Ks[cur][tRow][dCol]) = kd;
    }
#endif
    { // V staged transposed by all 256 threads (overlaps in-flight DMA)
      v8h vd = *(const v8h*)(vp + (size_t)(kv + tRow) * HDIM + dCol);
#pragma unroll
      for (int j = 0; j < 8; ++j) VsT[dCol + j][tRow] = vd[j];
    }
#if HAVE_TDM
    if (wave == 0) {
      if (kv + 32 < kvend) __builtin_amdgcn_s_wait_tensorcnt((short)1);
      else                 __builtin_amdgcn_s_wait_tensorcnt((short)0);
    }
#endif
    __syncthreads();           // publish Ks[cur] (DMA) and VsT (VALU stores)

    // scores 16x32 : two 16-col blocks, each K=64 -> two WMMA k-steps
    v8f s0, s1;
#pragma unroll
    for (int e = 0; e < 8; ++e) { s0[e] = 0.f; s1[e] = 0.f; }
    {
      v16h kb00 = load_b_frag(&Ks[cur][0][0],  80);
      v16h kb01 = load_b_frag(&Ks[cur][0][32], 80);
      v16h kb10 = load_b_frag(&Ks[cur][16][0], 80);
      v16h kb11 = load_b_frag(&Ks[cur][16][32],80);
      s0 = wmma_f16(qf0, kb00, s0); s0 = wmma_f16(qf1, kb01, s0);
      s1 = wmma_f16(qf0, kb10, s1); s1 = wmma_f16(qf1, kb11, s1);
    }

    // causal mask + online softmax (row = r + 8*hi, cols across 16 lanes)
    int qrow_base = q0 + 8 * hi;
    int t0 = kv + l15, t1 = kv + 16 + l15;
    float scale8[8];
#pragma unroll
    for (int r = 0; r < 8; ++r) {
      int qr = qrow_base + r;
      float a = (t0 <= qr) ? s0[r] : -__builtin_inff();
      float c = (t1 <= qr) ? s1[r] : -__builtin_inff();
      float rm = fmaxf(a, c);
#pragma unroll
      for (int m = 1; m < 16; m <<= 1) rm = fmaxf(rm, __shfl_xor(rm, m, 32));
      float mnew = fmaxf(mrun[r], rm);
      float p0 = __expf(a - mnew);
      float p1 = __expf(c - mnew);
      float rs = p0 + p1;
#pragma unroll
      for (int m = 1; m < 16; m <<= 1) rs += __shfl_xor(rs, m, 32);
      float sc = __expf(mrun[r] - mnew);
      lrun[r] = lrun[r] * sc + rs;
      mrun[r] = mnew;
      scale8[r] = sc;
      Pbuf[wave][r + 8 * hi][l15]      = (half_t)p0;
      Pbuf[wave][r + 8 * hi][16 + l15] = (half_t)p1;
    }
#pragma unroll
    for (int j = 0; j < 4; ++j)
#pragma unroll
      for (int r = 0; r < 8; ++r) oacc[j][r] *= scale8[r];

    // same-wave LDS RAW on Pbuf: wait for DS stores before fragment reload
    asm volatile("s_wait_dscnt 0" ::: "memory");

    v16h pf = load_a_frag(&Pbuf[wave][0][0], 32);
#pragma unroll
    for (int j = 0; j < 4; ++j) {
      v16h vb = load_b_frag(&VsT[j * 16][0], 40);
      oacc[j] = wmma_f16(pf, vb, oacc[j]);
    }

    cur ^= 1;
  }

  // epilogue: normalize and write [B,S,E] f16
#pragma unroll
  for (int r = 0; r < 8; ++r) {
    int q = q0 + 8 * hi + r;
    float inv = 1.0f / lrun[r];
#pragma unroll
    for (int j = 0; j < 4; ++j) {
      int d = j * 16 + l15;
      size_t idx = (((size_t)b * SEQ) + q) * EMBED + h * HDIM + d;
      O[idx] = (half_t)(oacc[j][r] * inv);
    }
  }
}

// ---------------------------------------------------------------------------
// Host launcher
// ---------------------------------------------------------------------------
extern "C" void kernel_launch(void* const* d_in, const int* in_sizes, int n_in,
                              void* d_out, int out_size, void* d_ws, size_t ws_size,
                              hipStream_t stream) {
  const float* x  = (const float*)d_in[0];
  const float* Wq = (const float*)d_in[1];
  const float* bq = (const float*)d_in[2];
  const float* Wk = (const float*)d_in[3];
  const float* bk = (const float*)d_in[4];
  const float* Wv = (const float*)d_in[5];
  const float* bv = (const float*)d_in[6];
  const float* Wo = (const float*)d_in[7];
  const float* bo = (const float*)d_in[8];
  float* out = (float*)d_out;

  half_t* p    = (half_t*)d_ws;
  half_t* x16  = p; p += (size_t)MTOT * EMBED;
  half_t* wq16 = p; p += (size_t)EMBED * EMBED;
  half_t* wk16 = p; p += (size_t)EMBED * EMBED;
  half_t* wv16 = p; p += (size_t)EMBED * EMBED;
  half_t* wo16 = p; p += (size_t)EMBED * EMBED;
  half_t* q16  = p; p += (size_t)BH * SEQ * HDIM;
  half_t* k16  = p; p += (size_t)BH * SEQ * HDIM;
  half_t* v16  = p; p += (size_t)BH * SEQ * HDIM;
  half_t* a16  = p; p += (size_t)MTOT * EMBED;

  k_f32_to_f16<<<1024, 256, 0, stream>>>(x,  x16,  MTOT * EMBED);
  k_f32_to_f16<<<512,  256, 0, stream>>>(Wq, wq16, EMBED * EMBED);
  k_f32_to_f16<<<512,  256, 0, stream>>>(Wk, wk16, EMBED * EMBED);
  k_f32_to_f16<<<512,  256, 0, stream>>>(Wv, wv16, EMBED * EMBED);
  k_f32_to_f16<<<512,  256, 0, stream>>>(Wo, wo16, EMBED * EMBED);

  dim3 g(EMBED / 128, MTOT / 256);   // (8, 32)
  k_proj_qkv<<<g, 256, 0, stream>>>(x16, wq16, bq, q16, 0.125f);  // 1/sqrt(64)
  k_proj_qkv<<<g, 256, 0, stream>>>(x16, wk16, bk, k16, 1.0f);
  k_proj_qkv<<<g, 256, 0, stream>>>(x16, wv16, bv, v16, 1.0f);

  k_attn<<<dim3(SEQ / 128, BH), 256, 0, stream>>>(q16, k16, v16, a16);

  k_proj_out<<<g, 256, 0, stream>>>(a16, wo16, bo, out);
}
